// _EncoderAttention_55095840473764
// MI455X (gfx1250) — matrix operand order repaired and checked by reading, not compiled, not run on hardware
//
#include <hip/hip_runtime.h>
#include <hip/hip_bf16.h>
#include <math.h>

typedef _Float16 f16;
typedef __attribute__((ext_vector_type(16))) _Float16 v16h;
typedef __attribute__((ext_vector_type(8)))  float    v8f;

#define B_    2
#define L_    2048
#define H_    16
#define KV_   4
#define D_    128
#define HID_  2048
#define NREP_ (H_ / KV_)

#define WMMA(a, b, c) \
  __builtin_amdgcn_wmma_f32_16x16x32_f16(false, (a), false, (b), (short)0, (c), false, false)

// Async global->LDS path (CDNA5): guarded so missing builtins fall back cleanly.
#if defined(__HIP_DEVICE_COMPILE__) && __has_builtin(__builtin_amdgcn_global_load_async_to_lds_b128)
#define USE_ASYNC 1
#else
#define USE_ASYNC 0
#endif

#if USE_ASYNC
typedef int v4i __attribute__((vector_size(16)));
typedef __attribute__((address_space(1))) v4i* gv4i_p;   // global (AS1) v4i*
typedef __attribute__((address_space(3))) v4i* lv4i_p;   // LDS   (AS3) v4i*
static __device__ __forceinline__ void async_cp_b128(const f16* g, f16* l) {
  __builtin_amdgcn_global_load_async_to_lds_b128((gv4i_p)g, (lv4i_p)l, 0, 0);
}
static __device__ __forceinline__ void wait_async0() {
#if __has_builtin(__builtin_amdgcn_s_wait_asynccnt)
  __builtin_amdgcn_s_wait_asynccnt(0);
#else
  asm volatile("s_wait_asynccnt 0x0" ::: "memory");
#endif
}
#endif

// ---------------------------------------------------------------------------
// WMMA fragment loaders matching the CDNA5 ISA 16-bit layouts (wave32).
// A 16x32 (MxK): lane&15 = M row; elems 0..7 -> K=8*(lane>>4)+e, 8..15 -> +16.
// B 32x16 (KxN) from an N-major array: lane&15 = N; elems 0..7 -> K=8*(lane>>4)+e, 8..15 -> +16.
// ---------------------------------------------------------------------------
static __device__ __forceinline__ v16h wmma_a_frag(const f16* p0, int ld, int lane) {
  const int m  = lane & 15;
  const int kh = (lane >> 4) << 3;
  const f16* p = p0 + (size_t)m * ld + kh;
  v16h a;
#pragma unroll
  for (int e = 0; e < 8; ++e) a[e] = p[e];
#pragma unroll
  for (int e = 0; e < 8; ++e) a[8 + e] = p[16 + e];
  return a;
}

static __device__ __forceinline__ v16h wmma_b_frag_nmajor(const f16* p0, size_t ld, int lane) {
  const int n  = lane & 15;
  const int kh = (lane >> 4) << 3;
  const f16* p = p0 + (size_t)n * ld + kh;
  v16h b;
#pragma unroll
  for (int e = 0; e < 8; ++e) b[e] = p[e];
#pragma unroll
  for (int e = 0; e < 8; ++e) b[8 + e] = p[16 + e];
  return b;
}

// ---------------------------------------------------------------------------
// fp32 -> f16 convert (straight)
// ---------------------------------------------------------------------------
__global__ __launch_bounds__(256) void cvt_f32_f16(const float* __restrict__ x,
                                                   f16* __restrict__ y, long n) {
  long i = (long)blockIdx.x * 256 + threadIdx.x;
  if (i < n) y[i] = (f16)x[i];
}

// ---------------------------------------------------------------------------
// fp32 (K,N) -> f16 (N,K) transpose-convert, 32x32 LDS tile (one-time on weights)
// grid: (N/32, K/32), 256 threads
// ---------------------------------------------------------------------------
__global__ __launch_bounds__(256) void cvt_transpose(const float* __restrict__ x,
                                                     f16* __restrict__ y,
                                                     int Kdim, int Ndim) {
  __shared__ float t[32][33];
  const int bx = blockIdx.x * 32;   // N
  const int by = blockIdx.y * 32;   // K
  const int tx = threadIdx.x & 31;
  const int ty = threadIdx.x >> 5;  // 0..7
#pragma unroll
  for (int i = 0; i < 32; i += 8)
    t[ty + i][tx] = x[(size_t)(by + ty + i) * Ndim + bx + tx];
  __syncthreads();
#pragma unroll
  for (int i = 0; i < 32; i += 8)
    y[(size_t)(bx + ty + i) * Kdim + by + tx] = (f16)t[tx][ty + i];
}

// ---------------------------------------------------------------------------
// Tiled WMMA GEMM: C[M,N](f32) = A[M,K](f16) @ Bt[N,K](f16, pre-transposed)
// 128x128 block, BK=64, 256 threads = 8 waves, wave tile 32x64 (2x4 accums).
// Double-buffered LDS; async global->LDS staging when available.
// ---------------------------------------------------------------------------
#define GM 128
#define GN 128
#define GK 64
#define GP 8
__global__ __launch_bounds__(256) void gemm_f16(const f16* __restrict__ A,
                                                const f16* __restrict__ Bt,
                                                float* __restrict__ C,
                                                int Kdim, int Ndim) {
  __shared__ __align__(32) f16 As[2][GM][GK + GP];   // 2 x 18 KB
  __shared__ __align__(32) f16 Bs[2][GN][GK + GP];   // 2 x 18 KB (N-major)

  const int tid  = threadIdx.x;
  const int lane = tid & 31;
  const int wave = tid >> 5;
  const int bm = blockIdx.y * GM;
  const int bn = blockIdx.x * GN;
  const int wr = (wave >> 1) * 32;
  const int wc = (wave & 1) * 64;

  // staging map: row = tid>>1, 32-half (64B) chunk = (tid&1)*32
  const int sr = tid >> 1, sc = (tid & 1) * 32;
  const f16* ga = A  + (size_t)(bm + sr) * Kdim + sc;
  const f16* gb = Bt + (size_t)(bn + sr) * Kdim + sc;

  const v8f vzero = {};
  v8f acc[2][4];
#pragma unroll
  for (int i = 0; i < 2; ++i)
#pragma unroll
    for (int j = 0; j < 4; ++j) acc[i][j] = vzero;

#if USE_ASYNC
  // prologue: async-stage tile 0 into buffer 0
#pragma unroll
  for (int c = 0; c < 4; ++c) {
    async_cp_b128(ga + c * 8, &As[0][sr][sc + c * 8]);
    async_cp_b128(gb + c * 8, &Bs[0][sr][sc + c * 8]);
  }
  wait_async0();
  __syncthreads();

  int buf = 0;
  for (int k0 = 0; k0 < Kdim; k0 += GK) {
    if (k0 + GK < Kdim) {  // async-stage next tile into the other buffer
#pragma unroll
      for (int c = 0; c < 4; ++c) {
        async_cp_b128(ga + k0 + GK + c * 8, &As[buf ^ 1][sr][sc + c * 8]);
        async_cp_b128(gb + k0 + GK + c * 8, &Bs[buf ^ 1][sr][sc + c * 8]);
      }
    }
#pragma unroll
    for (int kk = 0; kk < 2; ++kk) {
      v16h af[2], bf[4];
#pragma unroll
      for (int i = 0; i < 2; ++i) af[i] = wmma_a_frag(&As[buf][wr + i * 16][kk * 32], GK + GP, lane);
#pragma unroll
      for (int j = 0; j < 4; ++j) bf[j] = wmma_b_frag_nmajor(&Bs[buf][wc + j * 16][kk * 32], GK + GP, lane);
#pragma unroll
      for (int i = 0; i < 2; ++i)
#pragma unroll
        for (int j = 0; j < 4; ++j) acc[i][j] = WMMA(af[i], bf[j], acc[i][j]);
    }
    wait_async0();      // our async writes into buf^1 are done
    __syncthreads();    // all waves done reading buf / writing buf^1
    buf ^= 1;
  }
#else
  // fallback: single-buffer sync staging (b128 both sides)
  for (int k0 = 0; k0 < Kdim; k0 += GK) {
    v16h ra0 = *(const v16h*)(ga + k0);
    v16h ra1 = *(const v16h*)(ga + k0 + 16);
    v16h rb0 = *(const v16h*)(gb + k0);
    v16h rb1 = *(const v16h*)(gb + k0 + 16);
    if (k0 + GK < Kdim) {
      __builtin_prefetch(ga + k0 + GK, 0, 1);
      __builtin_prefetch(gb + k0 + GK, 0, 1);
    }
    __syncthreads();   // WAR vs previous compute
    *(v16h*)&As[0][sr][sc]      = ra0;
    *(v16h*)&As[0][sr][sc + 16] = ra1;
    *(v16h*)&Bs[0][sr][sc]      = rb0;
    *(v16h*)&Bs[0][sr][sc + 16] = rb1;
    __syncthreads();
#pragma unroll
    for (int kk = 0; kk < 2; ++kk) {
      v16h af[2], bf[4];
#pragma unroll
      for (int i = 0; i < 2; ++i) af[i] = wmma_a_frag(&As[0][wr + i * 16][kk * 32], GK + GP, lane);
#pragma unroll
      for (int j = 0; j < 4; ++j) bf[j] = wmma_b_frag_nmajor(&Bs[0][wc + j * 16][kk * 32], GK + GP, lane);
#pragma unroll
      for (int i = 0; i < 2; ++i)
#pragma unroll
        for (int j = 0; j < 4; ++j) acc[i][j] = WMMA(af[i], bf[j], acc[i][j]);
    }
  }
#endif

  // C fragment layout: VGPR r -> row r (lanes 0..15) / 8+r (lanes 16..31); lane&15 = col
  const int cn = lane & 15;
  const int rb = (lane >> 4) * 8;
#pragma unroll
  for (int i = 0; i < 2; ++i)
#pragma unroll
    for (int j = 0; j < 4; ++j)
#pragma unroll
      for (int r = 0; r < 8; ++r)
        C[(size_t)(bm + wr + i * 16 + rb + r) * Ndim + bn + wc + j * 16 + cn] = acc[i][j][r];
}

// ---------------------------------------------------------------------------
// Per-head RMS-norm + RoPE. One wave per (b,l,head) row of D=128.
// Output layout: Y[(b*nh + h)*L + l][D]
// ---------------------------------------------------------------------------
__global__ __launch_bounds__(128) void rope_norm(const float* __restrict__ X,
                                                 const float* __restrict__ w,
                                                 const float* __restrict__ cosb,
                                                 const float* __restrict__ sinb,
                                                 f16* __restrict__ Y, int nh) {
  const int lane = threadIdx.x & 31;
  const int wave = threadIdx.x >> 5;
  const int rid  = blockIdx.x * 4 + wave;
  const int bl   = rid / nh;
  const int hh   = rid - bl * nh;
  const float* x = X + (size_t)bl * (nh * D_) + (size_t)hh * D_;

  float v[4], vn[4];
  float ss = 0.f;
#pragma unroll
  for (int i = 0; i < 4; ++i) { v[i] = x[lane * 4 + i]; ss += v[i] * v[i]; }
#pragma unroll
  for (int m = 16; m >= 1; m >>= 1) ss += __shfl_xor(ss, m, 32);
  const float rs = rsqrtf(ss * (1.0f / D_) + 1e-6f);
#pragma unroll
  for (int i = 0; i < 4; ++i) vn[i] = v[i] * rs * w[lane * 4 + i];

  const float sgn = (lane < 16) ? -1.0f : 1.0f;   // rotate_half: d^64 <-> lane^16
  const int b = bl / L_;
  const int l = bl - b * L_;
  f16* y = Y + ((size_t)(b * nh + hh) * L_ + l) * D_;
#pragma unroll
  for (int i = 0; i < 4; ++i) {
    const int d = lane * 4 + i;
    const float rh = sgn * __shfl_xor(vn[i], 16, 32);
    const float c  = cosb[(size_t)bl * D_ + d];
    const float s  = sinb[(size_t)bl * D_ + d];
    y[d] = (f16)(vn[i] * c + rh * s);
  }
}

// ---------------------------------------------------------------------------
// V transpose: (B*L, KV*D) f32 -> Vt (B,KV,D,L) f16
// ---------------------------------------------------------------------------
__global__ __launch_bounds__(256) void v_transpose(const float* __restrict__ V,
                                                   f16* __restrict__ Vt) {
  const long i = (long)blockIdx.x * 256 + threadIdx.x;
  const int bl = (int)(i >> 9);
  const int r  = (int)(i & 511);
  const int kv = r >> 7;
  const int d  = r & 127;
  const int b  = bl / L_;
  const int l  = bl - b * L_;
  Vt[((size_t)(b * KV_ + kv) * D_ + d) * L_ + l] = (f16)V[i];
}

// ---------------------------------------------------------------------------
// Flash attention (non-causal). Grid: (L/64, B*H), 128 threads = 4 waves.
// Q: (B*H, L, D)  K: (B*KV, L, D)  Vt: (B*KV, D, L)  O: (B*L, H*D), all f16.
// ---------------------------------------------------------------------------
#define ABM 64
#define ABN 64
__global__ __launch_bounds__(128) void attn(const f16* __restrict__ Q,
                                            const f16* __restrict__ Kc,
                                            const f16* __restrict__ Vt,
                                            f16* __restrict__ O) {
  const int lane = threadIdx.x & 31;
  const int wave = threadIdx.x >> 5;
  const int bh = blockIdx.y;
  const int b  = bh / H_;
  const int h  = bh - b * H_;
  const int g  = h / NREP_;
  const int q0 = blockIdx.x * ABM + wave * 16;

  const f16* Qb = Q  + ((size_t)bh * L_ + q0) * D_;
  const f16* Kb = Kc + (size_t)(b * KV_ + g) * L_ * D_;
  const f16* Vb = Vt + (size_t)(b * KV_ + g) * D_ * L_;

  __shared__ __align__(32) f16 Ps[4][16][ABN + 8];   // per-wave P relayout tile

  v16h qf[4];
#pragma unroll
  for (int kk = 0; kk < 4; ++kk) qf[kk] = wmma_a_frag(Qb + kk * 32, D_, lane);

  const v8f vzero = {};
  v8f o[8];
#pragma unroll
  for (int d = 0; d < 8; ++d) o[d] = vzero;
  float mrow[8], lrow[8];
#pragma unroll
  for (int r = 0; r < 8; ++r) { mrow[r] = -1e30f; lrow[r] = 0.f; }

  const float scale = 0.08838834764831845f;  // 1/sqrt(128)
  const int cn = lane & 15;
  const int rb = (lane >> 4) * 8;

  for (int kb = 0; kb < L_; kb += ABN) {
    v8f s[4];
#pragma unroll
    for (int j = 0; j < 4; ++j) s[j] = vzero;
#pragma unroll
    for (int j = 0; j < 4; ++j) {
      const f16* kp = Kb + (size_t)(kb + j * 16) * D_;
#pragma unroll
      for (int kk = 0; kk < 4; ++kk) {
        v16h bf = wmma_b_frag_nmajor(kp + kk * 32, D_, lane);
        s[j] = WMMA(qf[kk], bf, s[j]);
      }
    }

    // online softmax: row r lives at (VGPR r, lane half); reduce across 16 lanes
#pragma unroll
    for (int r = 0; r < 8; ++r) {
      float mx = -1e30f;
#pragma unroll
      for (int j = 0; j < 4; ++j) { s[j][r] *= scale; mx = fmaxf(mx, s[j][r]); }
#pragma unroll
      for (int m = 8; m >= 1; m >>= 1) mx = fmaxf(mx, __shfl_xor(mx, m, 32));
      const float mnew = fmaxf(mrow[r], mx);
      const float corr = __expf(mrow[r] - mnew);
      float rsum = 0.f;
#pragma unroll
      for (int j = 0; j < 4; ++j) { const float p = __expf(s[j][r] - mnew); s[j][r] = p; rsum += p; }
#pragma unroll
      for (int m = 8; m >= 1; m >>= 1) rsum += __shfl_xor(rsum, m, 32);
      lrow[r] = lrow[r] * corr + rsum;
      mrow[r] = mnew;
#pragma unroll
      for (int d = 0; d < 8; ++d) o[d][r] *= corr;
    }

    // relayout P (C-frag -> A-frag) through per-wave LDS tile
#pragma unroll
    for (int j = 0; j < 4; ++j)
#pragma unroll
      for (int r = 0; r < 8; ++r)
        Ps[wave][rb + r][j * 16 + cn] = (f16)s[j][r];
    asm volatile("s_wait_dscnt 0" ::: "memory");

    // O += P @ V
#pragma unroll
    for (int kk = 0; kk < 2; ++kk) {
      v16h pf = wmma_a_frag(&Ps[wave][0][kk * 32], ABN + 8, lane);
#pragma unroll
      for (int d = 0; d < 8; ++d) {
        v16h vf = wmma_b_frag_nmajor(Vb + (size_t)(d * 16) * L_ + kb + kk * 32, L_, lane);
        o[d] = WMMA(pf, vf, o[d]);
      }
    }
    asm volatile("s_wait_dscnt 0" ::: "memory");   // WAR vs next iteration's stores
  }

  float inv[8];
#pragma unroll
  for (int r = 0; r < 8; ++r) inv[r] = 1.0f / lrow[r];
  f16* ob = O + ((size_t)(b * L_) + q0) * (H_ * D_) + (size_t)h * D_;
#pragma unroll
  for (int d = 0; d < 8; ++d)
#pragma unroll
    for (int r = 0; r < 8; ++r)
      ob[(size_t)(rb + r) * (H_ * D_) + d * 16 + cn] = (f16)(o[d][r] * inv[r]);
}

// ---------------------------------------------------------------------------
// Host launch
// ---------------------------------------------------------------------------
extern "C" void kernel_launch(void* const* d_in, const int* in_sizes, int n_in,
                              void* d_out, int out_size, void* d_ws, size_t ws_size,
                              hipStream_t stream) {
  (void)in_sizes; (void)n_in; (void)out_size; (void)ws_size;

  const float* hs   = (const float*)d_in[0];
  const float* cosb = (const float*)d_in[1];
  const float* sinb = (const float*)d_in[2];
  const float* Wq   = (const float*)d_in[3];
  const float* Wk   = (const float*)d_in[4];
  const float* Wv   = (const float*)d_in[5];
  const float* Wo   = (const float*)d_in[6];
  const float* qw   = (const float*)d_in[7];
  const float* kw   = (const float*)d_in[8];

  char* ws = (char*)d_ws;
  size_t off = 0;
  auto alloc = [&](size_t bytes) -> char* {
    char* p = ws + off;
    off += (bytes + 255) & ~(size_t)255;
    return p;
  };

  const size_t ML = (size_t)B_ * L_;  // 4096
  f16*   hs16 = (f16*)  alloc(ML * HID_ * 2);
  f16*   wq16 = (f16*)  alloc((size_t)HID_ * H_ * D_ * 2);    // transposed (N,K)
  f16*   wk16 = (f16*)  alloc((size_t)HID_ * KV_ * D_ * 2);   // transposed
  f16*   wv16 = (f16*)  alloc((size_t)HID_ * KV_ * D_ * 2);   // transposed
  f16*   wo16 = (f16*)  alloc((size_t)H_ * D_ * HID_ * 2);    // transposed
  float* qraw = (float*)alloc(ML * H_ * D_ * 4);
  float* kraw = (float*)alloc(ML * KV_ * D_ * 4);
  float* vraw = (float*)alloc(ML * KV_ * D_ * 4);
  f16*   q16  = (f16*)  alloc(ML * H_ * D_ * 2);
  f16*   k16  = (f16*)  alloc(ML * KV_ * D_ * 2);
  f16*   vt16 = (f16*)  alloc(ML * KV_ * D_ * 2);
  f16*   ao16 = (f16*)  alloc(ML * H_ * D_ * 2);

  cvt_f32_f16<<<(unsigned)((ML * HID_ + 255) / 256), 256, 0, stream>>>(hs, hs16, (long)(ML * HID_));

  // one-time transpose-convert of weights: (K,N) f32 -> (N,K) f16
  cvt_transpose<<<dim3((H_ * D_) / 32, HID_ / 32), 256, 0, stream>>>(Wq, wq16, HID_, H_ * D_);
  cvt_transpose<<<dim3((KV_ * D_) / 32, HID_ / 32), 256, 0, stream>>>(Wk, wk16, HID_, KV_ * D_);
  cvt_transpose<<<dim3((KV_ * D_) / 32, HID_ / 32), 256, 0, stream>>>(Wv, wv16, HID_, KV_ * D_);
  cvt_transpose<<<dim3(HID_ / 32, (H_ * D_) / 32), 256, 0, stream>>>(Wo, wo16, H_ * D_, HID_);

  // QKV projections
  gemm_f16<<<dim3((H_ * D_) / GN, (unsigned)(ML / GM)), 256, 0, stream>>>(hs16, wq16, qraw, HID_, H_ * D_);
  gemm_f16<<<dim3((KV_ * D_) / GN, (unsigned)(ML / GM)), 256, 0, stream>>>(hs16, wk16, kraw, HID_, KV_ * D_);
  gemm_f16<<<dim3((KV_ * D_) / GN, (unsigned)(ML / GM)), 256, 0, stream>>>(hs16, wv16, vraw, HID_, KV_ * D_);

  // norm + RoPE + layout
  rope_norm<<<(unsigned)(ML * H_ / 4), 128, 0, stream>>>(qraw, qw, cosb, sinb, q16, H_);
  rope_norm<<<(unsigned)(ML * KV_ / 4), 128, 0, stream>>>(kraw, kw, cosb, sinb, k16, KV_);
  v_transpose<<<(unsigned)(ML * KV_ * D_ / 256), 256, 0, stream>>>(vraw, vt16);

  // flash attention
  attn<<<dim3(L_ / ABM, B_ * H_), 128, 0, stream>>>(q16, k16, vt16, ao16);

  // output projection -> fp32 d_out
  gemm_f16<<<dim3(HID_ / GN, (unsigned)(ML / GM)), 256, 0, stream>>>(ao16, wo16, (float*)d_out, H_ * D_, HID_);
}
